// LSTMScratch_86758339379650
// MI455X (gfx1250) — compile-verified, matrix operand order checked by reading, not run on hardware
//
#include <hip/hip_runtime.h>
#include <hip/hip_bf16.h>
#include <math.h>

// ---------------------------------------------------------------------------
// LSTM on MI455X (gfx1250, wave32):
//   gates[t] = [x_t ; h_{t-1}] (64x2048, bf16)  @  W_cat (2048x4096, bf16)
// v_wmma_f32_16x16x32_bf16 with fp32 accumulate; pointwise sigmoid/tanh
// epilogue updates C (fp32) and scatters H into the WMMA A-fragment layout
// (bf16) for the next step (ping-pong buffered).
//
// Round-5 changes: branchless pipeline phases. The x->h segment switch for
// A-fragments is hoisted out of the stage code: x-phase loop (refills from
// X_sw), one static transition group, h-phase loop (refills from H). No
// per-stage branches -> longer scheduling regions, deeper load flight.
//
// Workspace layout (requires ws_size >= ~84.2 MB):
//   [0,              16 MiB)   W_sw  : W_cat pre-swizzled to B-fragment layout
//   [16 MiB,         80 MiB)   X_sw  : inputs pre-swizzled to A-fragment layout
//   [80 MiB,  +128 KiB)        H_sw0 : H fragment buffer (ping)
//   [.. ,     +128 KiB)        H_sw1 : H fragment buffer (pong)
//   [.. ,     +256 KiB)        Cbuf  : cell state fp32 [64][1024]
// ---------------------------------------------------------------------------

typedef __attribute__((ext_vector_type(16))) __bf16 v16bf;
typedef __attribute__((ext_vector_type(8)))  float  v8f;

#define T_DIM 512
#define B_DIM 64
#define NI    1024
#define NH    1024
#define NG    (4 * NH)        // 4096 gate columns
#define NT16  (NG / 16)       // 256 column tiles over all gates
#define NHT16 (NH / 16)       // 64 column tiles over NH
#define MT16  (B_DIM / 16)    // 4 row tiles
#define KTI   (NI / 32)       // 32 k-tiles for x part
#define KTH   (NH / 32)       // 32 k-tiles for h part
#define KTT   (KTI + KTH)     // 64 total k-tiles
#define FRAG  512             // elems per fragment: 32 lanes * 16 bf16

// ---- fragment index helpers (ISA 16-bit A/B layout, wave32) ---------------
// lane in [0,16): holds K = 0..7 (j=0..7) and K = 16..23 (j=8..15)
// lane in [16,32): holds K = 8..15 (j=0..7) and K = 24..31 (j=8..15)
__device__ __forceinline__ int kmap(int lane, int j) {
  int base = (lane >= 16) ? 8 : 0;
  return base + ((j < 8) ? j : (j + 8));
}

__device__ __forceinline__ v16bf ldfrag(const __bf16* p, long frag, int lane) {
  return *(const v16bf*)(p + (frag * 32 + lane) * 16);
}

__device__ __forceinline__ float sigmoidf_(float x) {
  return 1.0f / (1.0f + __expf(-x));
}

// ---- kernel 0: zero H_sw0 + C ---------------------------------------------
__global__ void lstm_init_kernel(__bf16* Hsw0, float* Cbuf) {
  int i = blockIdx.x * blockDim.x + threadIdx.x;
  if (i < B_DIM * NH) {
    Hsw0[i] = (__bf16)0.0f;
    Cbuf[i] = 0.0f;
  }
}

// ---- kernel 1: inputs fp32 [T,B,NI] -> X_sw bf16 A-fragment layout --------
__global__ void lstm_pack_x_kernel(const float* __restrict__ inp,
                                   __bf16* __restrict__ Xsw) {
  long tid  = (long)blockIdx.x * blockDim.x + threadIdx.x;
  int  lane = (int)(tid & 31);
  int  mt   = (int)((tid >> 5) & (MT16 - 1));
  int  kt   = (int)((tid >> 7) & (KTI - 1));
  long t    = tid >> 12;
  if (t >= T_DIM) return;

  int m = mt * 16 + (lane & 15);
  const float* row = inp + (t * B_DIM + m) * NI + kt * 32;

  __bf16 v[16];
#pragma unroll
  for (int j = 0; j < 16; ++j) v[j] = (__bf16)row[kmap(lane, j)];

  __bf16* dst = Xsw + ((t * KTI + kt) * MT16 + mt) * (long)FRAG + lane * 16;
  *(v16bf*)dst = *(const v16bf*)v;
}

// ---- kernel 2: pack W_cat -> W_sw bf16 B-fragment layout ------------------
__global__ void lstm_pack_w_kernel(const float* __restrict__ Wxf, const float* __restrict__ Whf,
                                   const float* __restrict__ Wxi, const float* __restrict__ Whi,
                                   const float* __restrict__ Wxc, const float* __restrict__ Whc,
                                   const float* __restrict__ Wxo, const float* __restrict__ Who,
                                   __bf16* __restrict__ Wsw) {
  long tid   = (long)blockIdx.x * blockDim.x + threadIdx.x;
  int  lane  = (int)(tid & 31);
  int  ntile = (int)((tid >> 5) & (NT16 - 1));
  int  ktg   = (int)(tid >> 13);
  if (ktg >= KTT) return;

  int c    = ntile * 16 + (lane & 15);
  int gate = c >> 10;          // 0=f 1=i 2=c 3=o
  int cg   = c & (NH - 1);

  const float* Wx = (gate == 0) ? Wxf : (gate == 1) ? Wxi : (gate == 2) ? Wxc : Wxo;
  const float* Wh = (gate == 0) ? Whf : (gate == 1) ? Whi : (gate == 2) ? Whc : Who;

  __bf16 v[16];
#pragma unroll
  for (int j = 0; j < 16; ++j) {
    int k = ktg * 32 + kmap(lane, j);
    float w = (k < NI) ? Wx[(long)k * NH + cg] : Wh[(long)(k - NI) * NH + cg];
    v[j] = (__bf16)w;
  }
  __bf16* dst = Wsw + ((long)(ktg * NT16 + ntile) * 32 + lane) * 16;
  *(v16bf*)dst = *(const v16bf*)v;
}

// ---- kernel 3: one LSTM time step -----------------------------------------
// grid = (64 nt, 4 mt) single-wave workgroups, block = 32 threads.
// Each wave: 16 rows x 16 NH-cols, all four gates, K = 2048.
__global__ void __launch_bounds__(32)
__attribute__((amdgpu_waves_per_eu(1)))
lstm_step_kernel(const __bf16* __restrict__ Xsw_t,   // this step's x fragments
                 const __bf16* __restrict__ Hrd,     // prev-H fragments (read)
                 __bf16* __restrict__ Hwr,           // new-H fragments (write)
                 const __bf16* __restrict__ Wsw,
                 const float* __restrict__ b_f, const float* __restrict__ b_i,
                 const float* __restrict__ b_c, const float* __restrict__ b_o,
                 float* __restrict__ Cbuf,
                 float* __restrict__ outT,            // d_out + t*B*NH
                 float* __restrict__ finalH,
                 float* __restrict__ finalC,
                 int is_last) {
  const int lane = threadIdx.x;          // 0..31
  const int nt   = blockIdx.x;           // 0..63  (NH column tile)
  const int mt   = blockIdx.y;           // 0..3   (row tile)

  v8f accF = {}, accI = {}, accC = {}, accO = {};

  // depth-4 pipeline slots; EVERY index below is a literal constant
  v16bf A[4], Bf[4], Bi[4], Bc[4], Bo[4];

  // prologue: tiles 0..3 (all in the x segment since KTI=32 > 3)
#pragma unroll
  for (int p = 0; p < 4; ++p) {
    A[p]  = ldfrag(Xsw_t, (long)p * MT16 + mt, lane);
    long kw = (long)p * NT16;
    Bf[p] = ldfrag(Wsw, kw + 0 * NHT16 + nt, lane);
    Bi[p] = ldfrag(Wsw, kw + 1 * NHT16 + nt, lane);
    Bc[p] = ldfrag(Wsw, kw + 2 * NHT16 + nt, lane);
    Bo[p] = ldfrag(Wsw, kw + 3 * NHT16 + nt, lane);
  }

  // one pipeline stage: consume slot `slot` (tile kb0+slot), refill it with
  // tile kn = kb0+slot+4 whose A-fragment comes from `ASEG` at tile index
  // (kn - AOFF). slot is always a literal; ASEG/AOFF are phase constants.
#define PIPE_STAGE(slot, ASEG, AOFF)                                           \
  {                                                                            \
    accF = __builtin_amdgcn_wmma_f32_16x16x32_bf16(false, A[slot], false,      \
                Bf[slot], (short)0, accF, false, false);                       \
    accI = __builtin_amdgcn_wmma_f32_16x16x32_bf16(false, A[slot], false,      \
                Bi[slot], (short)0, accI, false, false);                       \
    accC = __builtin_amdgcn_wmma_f32_16x16x32_bf16(false, A[slot], false,      \
                Bc[slot], (short)0, accC, false, false);                       \
    accO = __builtin_amdgcn_wmma_f32_16x16x32_bf16(false, A[slot], false,      \
                Bo[slot], (short)0, accO, false, false);                       \
    const int kn = kb0 + (slot) + 4;                                           \
    long kpb = (long)(kn + 4) * NT16; /* prefetch one group further ahead */   \
    __builtin_prefetch(Wsw + ((kpb + 0 * NHT16 + nt) * 32 + lane) * 16, 0, 1); \
    __builtin_prefetch(Wsw + ((kpb + 1 * NHT16 + nt) * 32 + lane) * 16, 0, 1); \
    __builtin_prefetch(Wsw + ((kpb + 2 * NHT16 + nt) * 32 + lane) * 16, 0, 1); \
    __builtin_prefetch(Wsw + ((kpb + 3 * NHT16 + nt) * 32 + lane) * 16, 0, 1); \
    A[slot] = ldfrag((ASEG), (long)(kn - (AOFF)) * MT16 + mt, lane);           \
    long kw = (long)kn * NT16;                                                 \
    Bf[slot] = ldfrag(Wsw, kw + 0 * NHT16 + nt, lane);                         \
    Bi[slot] = ldfrag(Wsw, kw + 1 * NHT16 + nt, lane);                         \
    Bc[slot] = ldfrag(Wsw, kw + 2 * NHT16 + nt, lane);                         \
    Bo[slot] = ldfrag(Wsw, kw + 3 * NHT16 + nt, lane);                         \
  }

  // x-phase: groups 0..6 -> consume tiles 0..27, refill 4..31 (all from X)
#pragma unroll 1
  for (int g = 0; g < 7; ++g) {
    const int kb0 = g * 4;
    PIPE_STAGE(0, Xsw_t, 0)
    PIPE_STAGE(1, Xsw_t, 0)
    PIPE_STAGE(2, Xsw_t, 0)
    PIPE_STAGE(3, Xsw_t, 0)
  }

  // transition group: consume tiles 28..31, refill 32..35 (from H)
  {
    const int kb0 = 28;
    PIPE_STAGE(0, Hrd, KTI)
    PIPE_STAGE(1, Hrd, KTI)
    PIPE_STAGE(2, Hrd, KTI)
    PIPE_STAGE(3, Hrd, KTI)
  }

  // h-phase: groups 8..14 -> consume tiles 32..59, refill 36..63 (all from H)
#pragma unroll 1
  for (int g = 8; g < 15; ++g) {
    const int kb0 = g * 4;
    PIPE_STAGE(0, Hrd, KTI)
    PIPE_STAGE(1, Hrd, KTI)
    PIPE_STAGE(2, Hrd, KTI)
    PIPE_STAGE(3, Hrd, KTI)
  }
#undef PIPE_STAGE

  // drain: tiles 60..63 (static slot indices, no refill)
#pragma unroll
  for (int p = 0; p < 4; ++p) {
    accF = __builtin_amdgcn_wmma_f32_16x16x32_bf16(false, A[p], false, Bf[p], (short)0, accF, false, false);
    accI = __builtin_amdgcn_wmma_f32_16x16x32_bf16(false, A[p], false, Bi[p], (short)0, accI, false, false);
    accC = __builtin_amdgcn_wmma_f32_16x16x32_bf16(false, A[p], false, Bc[p], (short)0, accC, false, false);
    accO = __builtin_amdgcn_wmma_f32_16x16x32_bf16(false, A[p], false, Bo[p], (short)0, accO, false, false);
  }

  // --- pointwise epilogue ---
  // C/D fragment layout: VGPR r, lanes 0..15 -> M = r, lanes 16..31 -> M = r+8; N = lane & 15
  const int ncol = nt * 16 + (lane & 15);
  const float bf = b_f[ncol], bi = b_i[ncol], bc = b_c[ncol], bo = b_o[ncol];

  const int kt2 = ncol >> 5;
  const int kl  = ncol & 31;
  const int hi  = (kl >> 3) & 1;
  const int jj  = (kl & 7) | ((kl & 16) >> 1);

#pragma unroll
  for (int r = 0; r < 8; ++r) {
    int   m  = mt * 16 + r + ((lane >> 4) << 3);
    float f  = sigmoidf_(accF[r] + bf);
    float i  = sigmoidf_(accI[r] + bi);
    float g  = tanhf(accC[r] + bc);
    float o  = sigmoidf_(accO[r] + bo);

    long  ci = (long)m * NH + ncol;
    float cn = f * Cbuf[ci] + i * g;
    Cbuf[ci] = cn;
    float h  = o * tanhf(cn);

    outT[ci] = h;
    // scatter new H into A-fragment layout for next step
    int laneD = (m & 15) | (hi << 4);
    Hwr[((long)(kt2 * MT16 + mt) * 32 + laneD) * 16 + jj] = (__bf16)h;

    if (is_last) { finalH[ci] = h; finalC[ci] = cn; }
  }
}

extern "C" void kernel_launch(void* const* d_in, const int* in_sizes, int n_in,
                              void* d_out, int out_size, void* d_ws, size_t ws_size,
                              hipStream_t stream) {
  const float* inputs = (const float*)d_in[0];
  const float* Wxf = (const float*)d_in[1];  const float* Whf = (const float*)d_in[2];
  const float* bF  = (const float*)d_in[3];
  const float* Wxi = (const float*)d_in[4];  const float* Whi = (const float*)d_in[5];
  const float* bI  = (const float*)d_in[6];
  const float* Wxc = (const float*)d_in[7];  const float* Whc = (const float*)d_in[8];
  const float* bC  = (const float*)d_in[9];
  const float* Wxo = (const float*)d_in[10]; const float* Who = (const float*)d_in[11];
  const float* bO  = (const float*)d_in[12];

  float* out = (float*)d_out;
  float* finalH = out + (long)T_DIM * B_DIM * NH;
  float* finalC = finalH + (long)B_DIM * NH;

  char* ws = (char*)d_ws;
  __bf16* Wsw  = (__bf16*)(ws);                               // 16 MiB
  __bf16* Xsw  = (__bf16*)(ws + (16l << 20));                 // 64 MiB
  __bf16* Hsw0 = (__bf16*)(ws + (80l << 20));                 // 128 KiB
  __bf16* Hsw1 = (__bf16*)(ws + (80l << 20) + (128l << 10));  // 128 KiB
  float*  Cbuf = (float*)(ws + (80l << 20) + (256l << 10));   // 256 KiB

  // init H(ping) and C to zero
  lstm_init_kernel<<<(B_DIM * NH + 255) / 256, 256, 0, stream>>>(Hsw0, Cbuf);

  // pack inputs -> A-fragment layout (T*32*4*32 lanes)
  {
    long threads = (long)T_DIM * KTI * MT16 * 32;
    lstm_pack_x_kernel<<<(int)(threads / 256), 256, 0, stream>>>(inputs, Xsw);
  }
  // pack weights -> B-fragment layout (64*256*32 lanes)
  {
    long threads = (long)KTT * NT16 * 32;
    lstm_pack_w_kernel<<<(int)(threads / 256), 256, 0, stream>>>(
        Wxf, Whf, Wxi, Whi, Wxc, Whc, Wxo, Who, Wsw);
  }

  // sequential recurrence: 512 stream-ordered step launches
  dim3 grid(NHT16, MT16);  // 64 x 4 single-wave workgroups
  for (int t = 0; t < T_DIM; ++t) {
    const __bf16* Xt  = Xsw + (long)t * KTI * MT16 * FRAG;
    const __bf16* Hrd = (t & 1) ? Hsw1 : Hsw0;
    __bf16*       Hwr = (t & 1) ? Hsw0 : Hsw1;
    lstm_step_kernel<<<grid, 32, 0, stream>>>(
        Xt, Hrd, Hwr, Wsw, bF, bI, bC, bO, Cbuf,
        out + (long)t * B_DIM * NH, finalH, finalC, (t == T_DIM - 1) ? 1 : 0);
  }
}
